// Node_Embed_15745350107950
// MI455X (gfx1250) — matrix operand order repaired and checked by reading, not compiled, not run on hardware
//
#include <hip/hip_runtime.h>

// GCN-chain fused kernel for MI455X (gfx1250, wave32, WMMA).
// One workgroup per (b,t): everything LDS-resident in f16, f32 accumulate,
// only the final [64,128] f32 tile is stored to HBM (store-bound kernel).
// Roofline: output store = 268 MB @ 23.3 TB/s ~= 11.5 us floor; 34.4 GFLOP of
// matmul must run on WMMA f16 (f32 acc) to stay under that floor.

typedef _Float16 v16h __attribute__((ext_vector_type(16)));
typedef _Float16 v8h  __attribute__((ext_vector_type(8)));
typedef float    v8f  __attribute__((ext_vector_type(8)));

constexpr int kN = 64;    // nodes (S + L)
constexpr int kE = 64;    // embedding dim
constexpr int kO = 128;   // output dim
constexpr int kS = 48;    // state dim (scaled part)
constexpr int kP = 72;    // padded LDS row pitch (halves): 144B rows, bank-friendly

__device__ __forceinline__ v8f wmma_f16(v16h a, v16h b, v8f c) {
  // D = A(16x32 f16) x B(32x16 f16) + C(16x16 f32)
  return __builtin_amdgcn_wmma_f32_16x16x32_f16(
      /*neg_a=*/false, a, /*neg_b=*/false, b,
      /*c_mod=*/(short)0, c, /*reuse_a=*/false, /*reuse_b=*/false);
}

// A-operand fragment: M row-major [M][K] with pitch kP.
// ISA 16-bit A 16x32 layout: lanes 0-15 = rows M, halves 0..7 -> K=hi*8+0..7,
// halves 8..15 -> K=16+hi*8+0..7 (hi = lane>>4).
__device__ __forceinline__ v16h load_a_frag(const _Float16* M, int mb, int kb, int lane) {
  const int m  = mb + (lane & 15);
  const int hi = lane >> 4;
  const v8h lo = *(const v8h*)(M + m * kP + kb + hi * 8);
  const v8h hv = *(const v8h*)(M + m * kP + kb + 16 + hi * 8);
  v16h r;
#pragma unroll
  for (int i = 0; i < 8; ++i) { r[i] = lo[i]; r[i + 8] = hv[i]; }
  return r;
}

// B-operand fragment from a TRANSPOSED store Bt[N][K] (pitch kP).
// ISA 16-bit B 32x16 layout: lane n = lane&15, halves 0..7 -> K=hi*8+0..7, halves 8..15 -> K=16+hi*8+0..7.
__device__ __forceinline__ v16h load_b_frag(const _Float16* Bt, int nb, int kb, int lane) {
  const int n  = nb + (lane & 15);
  const int hi = lane >> 4;
  const v8h lo = *(const v8h*)(Bt + n * kP + kb + hi * 8);
  const v8h hv = *(const v8h*)(Bt + n * kP + kb + 16 + hi * 8);
  v16h r;
#pragma unroll
  for (int i = 0; i < 8; ++i) { r[i] = lo[i]; r[i + 8] = hv[i]; }
  return r;
}

// Full K=64 tile GEMM: D = A[mb:mb+16, :] x Bt[nb:nb+16, :]^T + C
__device__ __forceinline__ v8f gemm_tile(const _Float16* Amat, const _Float16* Bt,
                                         int mb, int nb, int lane, v8f c) {
  c = wmma_f16(load_a_frag(Amat, mb, 0,  lane), load_b_frag(Bt, nb, 0,  lane), c);
  c = wmma_f16(load_a_frag(Amat, mb, 32, lane), load_b_frag(Bt, nb, 32, lane), c);
  return c;
}

// C/D layout: lane holds column n = lane&15; VGPR r holds row m = r + 8*(lane>>4).
template <bool RELU, bool DUAL>
__device__ __forceinline__ void store_tile_f16(_Float16* dst, _Float16* dstT,
                                               int mb, int nb, int lane, v8f c) {
  const int n  = nb + (lane & 15);
  const int hi = lane >> 4;
#pragma unroll
  for (int r = 0; r < 8; ++r) {
    float v = c[r];
    if (RELU) v = fmaxf(v, 0.0f);
    const _Float16 h = (_Float16)v;
    const int m = mb + r + 8 * hi;
    dst[m * kP + n] = h;
    if (DUAL) dstT[n * kP + m] = h;
  }
}

__global__ __launch_bounds__(256) void gcn_fused_wmma(
    const float* __restrict__ x,      // [B*T, 48]
    const float* __restrict__ Wlift,  // [64, 64]   (EMB x N)
    const float* __restrict__ Wgcn,   // [3, 64, 64]
    const float* __restrict__ Wfp,    // [128, 64]
    const float* __restrict__ bfp,    // [128]
    const int*   __restrict__ A,      // [64, 64]
    float* __restrict__ out)          // [B*T, 64, 128]
{
  __shared__ alignas(32) _Float16 sAT [kN * kP];      // Ahat^T, A-operand
  __shared__ alignas(32) _Float16 sHA [kN * kP];      // h, [node][emb] (A-operand)
  __shared__ alignas(32) _Float16 sHB [kE * kP];      // h, [emb][node] (B-transposed)
  __shared__ alignas(32) _Float16 sTMP[kN * kP];      // Ahat^T @ h     (A-operand)
  __shared__ alignas(32) _Float16 sWG [3][kE * kP];   // W_gcn[l][f][e] (B-transposed)
  __shared__ alignas(32) _Float16 sWFP[kO * kP];      // W_fp[o][e]     (B-transposed)
  __shared__ float sInv[kN];

  const int tid = threadIdx.x;
  const int bt  = blockIdx.x;

  // --- column inv-norms of A (entries 0/1, self-loops => colsum >= 1) ---
  if (tid < kN) {
    int s = 0;
    for (int j = 0; j < kN; ++j) s += A[j * kN + tid];
    sInv[tid] = 1.0f / fmaxf(sqrtf((float)s), 1e-12f);
  }
  __syncthreads();

  // --- stage AhatT, h0 (both layouts), and f16 weights into LDS ---
  for (int idx = tid; idx < kN * kN; idx += 256) {
    const int r = idx >> 6, c = idx & 63;
    // AhatT[m][k] = Ahat[k][m] = A[k][m] * inv_colnorm[m]
    sAT[r * kP + c] = (_Float16)((float)A[c * kN + r] * sInv[r]);
    // h0[n][e] = scale[n] * W_lift[e][n]
    const float sc = (r < kS) ? x[bt * kS + r] : 1.0f;
    const float h0 = sc * Wlift[c * kN + r];
    sHA[r * kP + c] = (_Float16)h0;
    sHB[c * kP + r] = (_Float16)h0;
    sWG[0][r * kP + c] = (_Float16)Wgcn[idx];
    sWG[1][r * kP + c] = (_Float16)Wgcn[4096 + idx];
    sWG[2][r * kP + c] = (_Float16)Wgcn[8192 + idx];
  }
  for (int idx = tid; idx < kO * kE; idx += 256) {
    const int r = idx >> 6, c = idx & 63;
    sWFP[r * kP + c] = (_Float16)Wfp[idx];
  }
  __syncthreads();

  // Wave id as an SGPR so all tile indexing is scalar (no divergent loops).
  const int wave = __builtin_amdgcn_readfirstlane(tid >> 5);
  const int lane = tid & 31;

  // --- 3 GCN layers ---
  for (int l = 0; l < 3; ++l) {
    // tmp[i][e] = sum_j Ahat[j][i] * h[j][e]  == AhatT x h   (16 tiles, 2/wave)
#pragma unroll
    for (int ti = 0; ti < 2; ++ti) {
      const int t  = wave * 2 + ti;
      const int mb = (t >> 2) << 4, nb = (t & 3) << 4;
      v8f c = {};
      c = gemm_tile(sAT, sHB, mb, nb, lane, c);
      store_tile_f16<false, false>(sTMP, nullptr, mb, nb, lane, c);
    }
    __syncthreads();
    // h[n][f] = relu( sum_e tmp[n][e] * W_gcn[l][f][e] )
    const _Float16* W = sWG[l];
#pragma unroll
    for (int ti = 0; ti < 2; ++ti) {
      const int t  = wave * 2 + ti;
      const int mb = (t >> 2) << 4, nb = (t & 3) << 4;
      v8f c = {};
      c = gemm_tile(sTMP, W, mb, nb, lane, c);
      store_tile_f16<true, true>(sHA, sHB, mb, nb, lane, c);  // relu; both layouts
    }
    __syncthreads();
  }

  // --- final projection: out[n][o] = sum_e h[n][e]*W_fp[o][e] + b_fp[o] ---
  float* outp = out + (size_t)bt * (kN * kO);
#pragma unroll
  for (int ti = 0; ti < 4; ++ti) {
    const int t  = wave * 4 + ti;             // 32 tiles, 4/wave
    const int mb = (t >> 3) << 4;             // node tile
    const int ob = (t & 7) << 4;              // output tile
    const float bias = bfp[ob + (lane & 15)];
    v8f c;
#pragma unroll
    for (int r = 0; r < 8; ++r) c[r] = bias;
    c = gemm_tile(sHA, sWFP, mb, ob, lane, c);
    const int n  = ob + (lane & 15);
    const int hi = lane >> 4;
#pragma unroll
    for (int r = 0; r < 8; ++r) {
      const int m = mb + r + 8 * hi;
      outp[m * kO + n] = c[r];
    }
  }
}

extern "C" void kernel_launch(void* const* d_in, const int* in_sizes, int n_in,
                              void* d_out, int out_size, void* d_ws, size_t ws_size,
                              hipStream_t stream) {
  const float* x     = (const float*)d_in[0];  // [B,T,48]
  const float* Wlift = (const float*)d_in[1];  // [64,64]
  const float* Wgcn  = (const float*)d_in[2];  // [3,64,64]
  const float* Wfp   = (const float*)d_in[3];  // [128,64]
  const float* bfp   = (const float*)d_in[4];  // [128]
  const int*   A     = (const int*)d_in[5];    // [64,64]
  float*       out   = (float*)d_out;          // [B,T,64,128]

  const int bt_count = in_sizes[0] / kS;       // B*T = 8192
  gcn_fused_wmma<<<dim3(bt_count), dim3(256), 0, stream>>>(
      x, Wlift, Wgcn, Wfp, bfp, A, out);
}
